// GCN_Toppingetal_53472342835547
// MI455X (gfx1250) — matrix operand order repaired and checked by reading, not compiled, not run on hardware
//
#include <hip/hip_runtime.h>
#include <hip/hip_bf16.h>

// ---------------------------------------------------------------------------
// GCN (2-layer, GCNConv) for MI455X / gfx1250.
//   N=100000 nodes, E=3.2M edges, F_IN=512, HID=64, C=16
// GEMMs run on bf16 WMMA (v_wmma_f32_16x16x32_bf16) with f32 accumulation;
// scatter-adds are f32 L2 atomics (all hot buffers fit the 192MB L2).
// ---------------------------------------------------------------------------

#define GN  100000
#define GE  3200000
#define FIN 512
#define HID 64
#define CLS 16

typedef __attribute__((ext_vector_type(16))) __bf16 v16bf;
typedef __attribute__((ext_vector_type(8)))  __bf16 v8bf;
typedef __attribute__((ext_vector_type(4)))  __bf16 v4bf;
typedef __attribute__((ext_vector_type(8)))  float  v8f;
typedef __attribute__((ext_vector_type(4)))  float  v4f;

__device__ __forceinline__ void atomic_add_f32(float* p, float v) {
  __hip_atomic_fetch_add(p, v, __ATOMIC_RELAXED, __HIP_MEMORY_SCOPE_AGENT);
}

// ---------------------------------------------------------------------------
// Pre-convert W1 [512x64] and W2 [64x16] (f32, row-major K x N) into bf16
// fragment-ordered buffers matching the WMMA B-operand register layout:
//   B frag for (ntile, ktile): lane L holds column n = L%16,
//   K = ktile*32 + (j>>3)*16 + (L>>4)*8 + (j&7)  for j = 0..15.
// ---------------------------------------------------------------------------
__global__ void prep_weights_kernel(const float* __restrict__ W1,
                                    const float* __restrict__ W2,
                                    __bf16* __restrict__ W1f,
                                    __bf16* __restrict__ W2f) {
  int i = blockIdx.x * blockDim.x + threadIdx.x;
  if (i < (FIN / 32) * (HID / 16) * 32 * 16) {      // 4 ntiles * 16 ktiles * 32 lanes * 16
    int j    = i & 15;
    int lane = (i >> 4) & 31;
    int kt   = (i >> 9) & 15;
    int nt   = i >> 13;
    int k = kt * 32 + ((j >> 3) * 16) + ((lane >> 4) * 8) + (j & 7);
    int n = nt * 16 + (lane & 15);
    W1f[i] = (__bf16)W1[k * HID + n];
  }
  if (i < (HID / 32) * 32 * 16) {                   // 2 ktiles * 32 lanes * 16
    int j    = i & 15;
    int lane = (i >> 4) & 31;
    int kt   = i >> 9;
    int k = kt * 32 + ((j >> 3) * 16) + ((lane >> 4) * 8) + (j & 7);
    int n = lane & 15;
    W2f[i] = (__bf16)W2[k * CLS + n];
  }
}

// ---------------------------------------------------------------------------
// Weighted in-degree: deg[dst] += w[e]   (self-loop +1.0 folded into rsqrt).
// ---------------------------------------------------------------------------
__global__ void deg_kernel(const long long* __restrict__ ei,
                           const float* __restrict__ w,
                           float* __restrict__ deg) {
  int e = blockIdx.x * blockDim.x + threadIdx.x;
  if (e < GE) {
    int dst = (int)ei[(long long)GE + e];
    atomic_add_f32(&deg[dst], w[e]);
  }
}

__global__ void dinv_kernel(float* __restrict__ deg) {
  int i = blockIdx.x * blockDim.x + threadIdx.x;
  if (i < GN) {
    // self-loop weight 1.0 => degree always > 0
    deg[i] = rsqrtf(deg[i] + 1.0f);   // in place: deg -> dinv
  }
}

// ---------------------------------------------------------------------------
// GEMM1: t1[N,64] = x[N,512] @ W1.  One wave per 16x64 output tile.
// A frag (16x32 bf16): lane<16 -> row=lane,    K = {k0..k0+7, k0+16..k0+23}
//                      lane>=16 -> row=lane-16, K = {k0+8..k0+15, k0+24..k0+31}
// 16 k-steps * 4 n-tiles = 64 WMMAs per wave.
// ---------------------------------------------------------------------------
__global__ void __launch_bounds__(256)
gemm1_kernel(const float* __restrict__ x,
             const __bf16* __restrict__ W1f,
             float* __restrict__ t1) {
  int wave = (blockIdx.x * blockDim.x + threadIdx.x) >> 5;
  int lane = threadIdx.x & 31;
  if (wave >= GN / 16) return;                     // uniform per wave
  int m0    = wave * 16;
  int row   = m0 + (lane & 15);
  int khalf = (lane >> 4) * 8;
  const float* xrow = x + (size_t)row * FIN;
  const v16bf* B = (const v16bf*)W1f;              // 16 bf16 per fragment slot

  v8f acc0 = {}, acc1 = {}, acc2 = {}, acc3 = {};

  #pragma unroll 4
  for (int kt = 0; kt < FIN / 32; ++kt) {
    int k0 = kt * 32;
    v4f u0 = __builtin_nontemporal_load((const v4f*)(xrow + k0 + khalf));
    v4f u1 = __builtin_nontemporal_load((const v4f*)(xrow + k0 + khalf + 4));
    v4f u2 = __builtin_nontemporal_load((const v4f*)(xrow + k0 + 16 + khalf));
    v4f u3 = __builtin_nontemporal_load((const v4f*)(xrow + k0 + 16 + khalf + 4));
    v16bf a;
    a[0]  = (__bf16)u0.x; a[1]  = (__bf16)u0.y; a[2]  = (__bf16)u0.z; a[3]  = (__bf16)u0.w;
    a[4]  = (__bf16)u1.x; a[5]  = (__bf16)u1.y; a[6]  = (__bf16)u1.z; a[7]  = (__bf16)u1.w;
    a[8]  = (__bf16)u2.x; a[9]  = (__bf16)u2.y; a[10] = (__bf16)u2.z; a[11] = (__bf16)u2.w;
    a[12] = (__bf16)u3.x; a[13] = (__bf16)u3.y; a[14] = (__bf16)u3.z; a[15] = (__bf16)u3.w;

    v16bf b0 = B[(0 * 16 + kt) * 32 + lane];
    v16bf b1 = B[(1 * 16 + kt) * 32 + lane];
    v16bf b2 = B[(2 * 16 + kt) * 32 + lane];
    v16bf b3 = B[(3 * 16 + kt) * 32 + lane];

    acc0 = __builtin_amdgcn_wmma_f32_16x16x32_bf16(false, a, false, b0, (short)0, acc0, false, false);
    acc1 = __builtin_amdgcn_wmma_f32_16x16x32_bf16(false, a, false, b1, (short)0, acc1, false, false);
    acc2 = __builtin_amdgcn_wmma_f32_16x16x32_bf16(false, a, false, b2, (short)0, acc2, false, false);
    acc3 = __builtin_amdgcn_wmma_f32_16x16x32_bf16(false, a, false, b3, (short)0, acc3, false, false);
  }

  // C/D layout: VGPR v, lanes 0-15 -> M=v, lanes 16-31 -> M=v+8; N = lane%16
  int mbase = m0 + (lane >> 4) * 8;
  int n     = lane & 15;
  #pragma unroll
  for (int v = 0; v < 8; ++v) {
    float* r = t1 + (size_t)(mbase + v) * HID + n;
    r[0]  = acc0[v];
    r[16] = acc1[v];
    r[32] = acc2[v];
    r[48] = acc3[v];
  }
}

// ---------------------------------------------------------------------------
// Scatter layer 1: agg1[dst,:] += t1[src,:] * (dinv[src]*w*dinv[dst])
// 16 lanes per edge, float4 per lane. t1/agg1/dinv are L2-resident.
// ---------------------------------------------------------------------------
__global__ void __launch_bounds__(256)
scatter1_kernel(const long long* __restrict__ ei,
                const float* __restrict__ w,
                const float* __restrict__ dinv,
                const float* __restrict__ t1,
                float* __restrict__ agg1) {
  long long tid = (long long)blockIdx.x * blockDim.x + threadIdx.x;
  long long e   = tid >> 4;
  int f         = (int)(tid & 15) * 4;
  if (e >= GE) return;
  int src = (int)ei[e];
  int dst = (int)ei[(long long)GE + e];
  float norm = dinv[src] * w[e] * dinv[dst];
  v4f hv = *(const v4f*)(t1 + (size_t)src * HID + f);
  float* op = agg1 + (size_t)dst * HID + f;
  atomic_add_f32(op + 0, hv.x * norm);
  atomic_add_f32(op + 1, hv.y * norm);
  atomic_add_f32(op + 2, hv.z * norm);
  atomic_add_f32(op + 3, hv.w * norm);
}

// ---------------------------------------------------------------------------
// finalize1: h = relu(agg1 + t1*dinv^2 (self-loop) + b1), emitted as bf16.
// ---------------------------------------------------------------------------
__global__ void finalize1_kernel(const float* __restrict__ agg1,
                                 const float* __restrict__ t1,
                                 const float* __restrict__ dinv,
                                 const float* __restrict__ b1,
                                 __bf16* __restrict__ hbf) {
  int tid  = blockIdx.x * blockDim.x + threadIdx.x;
  int node = tid >> 4;
  int f    = (tid & 15) * 4;
  if (node >= GN) return;
  float di = dinv[node];
  float s  = di * di;
  v4f ag = *(const v4f*)(agg1 + (size_t)node * HID + f);
  v4f tv = *(const v4f*)(t1   + (size_t)node * HID + f);
  v4f bv = *(const v4f*)(b1 + f);
  float r0 = fmaxf(ag.x + tv.x * s + bv.x, 0.0f);
  float r1 = fmaxf(ag.y + tv.y * s + bv.y, 0.0f);
  float r2 = fmaxf(ag.z + tv.z * s + bv.z, 0.0f);
  float r3 = fmaxf(ag.w + tv.w * s + bv.w, 0.0f);
  v4bf o;
  o[0] = (__bf16)r0; o[1] = (__bf16)r1; o[2] = (__bf16)r2; o[3] = (__bf16)r3;
  *(v4bf*)(hbf + (size_t)node * HID + f) = o;
}

// ---------------------------------------------------------------------------
// GEMM2: t2[N,16] = h[N,64] @ W2.  One wave per 16x16 tile, K=64 -> 2 WMMAs.
// ---------------------------------------------------------------------------
__global__ void __launch_bounds__(256)
gemm2_kernel(const __bf16* __restrict__ hbf,
             const __bf16* __restrict__ W2f,
             float* __restrict__ t2) {
  int wave = (blockIdx.x * blockDim.x + threadIdx.x) >> 5;
  int lane = threadIdx.x & 31;
  if (wave >= GN / 16) return;
  int m0    = wave * 16;
  int row   = m0 + (lane & 15);
  int khalf = (lane >> 4) * 8;
  const __bf16* hrow = hbf + (size_t)row * HID;
  const v16bf* B = (const v16bf*)W2f;

  v8f acc = {};
  #pragma unroll
  for (int kt = 0; kt < HID / 32; ++kt) {
    v8bf lo = *(const v8bf*)(hrow + kt * 32 + khalf);
    v8bf hi = *(const v8bf*)(hrow + kt * 32 + 16 + khalf);
    v16bf a;
    #pragma unroll
    for (int j = 0; j < 8; ++j) { a[j] = lo[j]; a[j + 8] = hi[j]; }
    v16bf b = B[kt * 32 + lane];
    acc = __builtin_amdgcn_wmma_f32_16x16x32_bf16(false, a, false, b, (short)0, acc, false, false);
  }

  int mbase = m0 + (lane >> 4) * 8;
  int n     = lane & 15;
  #pragma unroll
  for (int v = 0; v < 8; ++v)
    t2[(size_t)(mbase + v) * CLS + n] = acc[v];
}

// ---------------------------------------------------------------------------
// Scatter layer 2: 4 lanes per edge, float4 per lane (16 features).
// ---------------------------------------------------------------------------
__global__ void __launch_bounds__(256)
scatter2_kernel(const long long* __restrict__ ei,
                const float* __restrict__ w,
                const float* __restrict__ dinv,
                const float* __restrict__ t2,
                float* __restrict__ agg2) {
  long long tid = (long long)blockIdx.x * blockDim.x + threadIdx.x;
  long long e   = tid >> 2;
  int f         = (int)(tid & 3) * 4;
  if (e >= GE) return;
  int src = (int)ei[e];
  int dst = (int)ei[(long long)GE + e];
  float norm = dinv[src] * w[e] * dinv[dst];
  v4f hv = *(const v4f*)(t2 + (size_t)src * CLS + f);
  float* op = agg2 + (size_t)dst * CLS + f;
  atomic_add_f32(op + 0, hv.x * norm);
  atomic_add_f32(op + 1, hv.y * norm);
  atomic_add_f32(op + 2, hv.z * norm);
  atomic_add_f32(op + 3, hv.w * norm);
}

// ---------------------------------------------------------------------------
// finalize2: out = log_softmax(agg2 + t2*dinv^2 + b2)  (row of 16 per thread)
// ---------------------------------------------------------------------------
__global__ void finalize2_kernel(const float* __restrict__ agg2,
                                 const float* __restrict__ t2,
                                 const float* __restrict__ dinv,
                                 const float* __restrict__ b2,
                                 float* __restrict__ out) {
  int i = blockIdx.x * blockDim.x + threadIdx.x;
  if (i >= GN) return;
  float di = dinv[i];
  float s  = di * di;
  float v[CLS];
  #pragma unroll
  for (int q = 0; q < CLS / 4; ++q) {
    v4f ag = *(const v4f*)(agg2 + (size_t)i * CLS + q * 4);
    v4f tv = *(const v4f*)(t2   + (size_t)i * CLS + q * 4);
    v4f bv = *(const v4f*)(b2 + q * 4);
    v[q * 4 + 0] = ag.x + tv.x * s + bv.x;
    v[q * 4 + 1] = ag.y + tv.y * s + bv.y;
    v[q * 4 + 2] = ag.z + tv.z * s + bv.z;
    v[q * 4 + 3] = ag.w + tv.w * s + bv.w;
  }
  float mx = v[0];
  #pragma unroll
  for (int q = 1; q < CLS; ++q) mx = fmaxf(mx, v[q]);
  float sum = 0.0f;
  #pragma unroll
  for (int q = 0; q < CLS; ++q) sum += __expf(v[q] - mx);
  float lse = mx + __logf(sum);
  #pragma unroll
  for (int q = 0; q < CLS; ++q) out[(size_t)i * CLS + q] = v[q] - lse;
}

// ---------------------------------------------------------------------------
// launch
// ---------------------------------------------------------------------------
extern "C" void kernel_launch(void* const* d_in, const int* in_sizes, int n_in,
                              void* d_out, int out_size, void* d_ws, size_t ws_size,
                              hipStream_t stream) {
  const float*     x  = (const float*)d_in[0];
  const long long* ei = (const long long*)d_in[1];   // int64 [2,E]
  const float*     ea = (const float*)d_in[2];
  const float*     W1 = (const float*)d_in[3];
  const float*     b1 = (const float*)d_in[4];
  const float*     W2 = (const float*)d_in[5];
  const float*     b2 = (const float*)d_in[6];
  float* out = (float*)d_out;

  // workspace layout (all offsets 32B-aligned)
  float* deg  = (float*)d_ws;                         // N      (becomes dinv)
  float* agg1 = deg  + GN;                            // N*64
  float* agg2 = agg1 + (size_t)GN * HID;              // N*16
  float* t1   = agg2 + (size_t)GN * CLS;              // N*64
  float* t2   = t1   + (size_t)GN * HID;              // N*16
  __bf16* hbf = (__bf16*)(t2 + (size_t)GN * CLS);     // N*64 bf16
  __bf16* W1f = hbf + (size_t)GN * HID;               // 32768 bf16
  __bf16* W2f = W1f + (FIN / 32) * (HID / 16) * 32 * 16;

  // zero deg + agg1 + agg2 in one shot (they are contiguous)
  size_t zbytes = sizeof(float) * ((size_t)GN * (1 + HID + CLS));
  (void)hipMemsetAsync(deg, 0, zbytes, stream);

  prep_weights_kernel<<<( (FIN/32)*(HID/16)*32*16 + 255) / 256, 256, 0, stream>>>(W1, W2, W1f, W2f);
  deg_kernel<<<(GE + 255) / 256, 256, 0, stream>>>(ei, ea, deg);
  dinv_kernel<<<(GN + 255) / 256, 256, 0, stream>>>(deg);

  gemm1_kernel<<<((GN / 16) * 32 + 255) / 256, 256, 0, stream>>>(x, W1f, t1);
  scatter1_kernel<<<(int)(((long long)GE * 16 + 255) / 256), 256, 0, stream>>>(ei, ea, deg, t1, agg1);
  finalize1_kernel<<<(GN * 16 + 255) / 256, 256, 0, stream>>>(agg1, t1, deg, b1, hbf);

  gemm2_kernel<<<((GN / 16) * 32 + 255) / 256, 256, 0, stream>>>(hbf, W2f, t2);
  scatter2_kernel<<<(int)(((long long)GE * 4 + 255) / 256), 256, 0, stream>>>(ei, ea, deg, t2, agg2);
  finalize2_kernel<<<(GN + 255) / 256, 256, 0, stream>>>(agg2, t2, deg, b2, out);
}
